// Seq2SeqModel_86105504350625
// MI455X (gfx1250) — matrix operand order repaired and checked by reading, not compiled, not run on hardware
//
#include <hip/hip_runtime.h>
#include <hip/hip_bf16.h>

// ---------------------------------------------------------------------------
// CDNA5 (gfx1250) WMMA seq2seq implementation.
// All matmuls use V_WMMA_F32_16X16X32_BF16 (M=16 == batch).
// ---------------------------------------------------------------------------

typedef __attribute__((ext_vector_type(16))) __bf16 v16bf;
typedef __attribute__((ext_vector_type(8)))  __bf16 v8bf;
typedef __attribute__((ext_vector_type(8)))  float  v8f;

#define DEV __device__ __forceinline__

DEV v8f wmma_bf16(v16bf a, v16bf b, v8f c) {
  // (neg_a, A, neg_b, B, c_mod, C, reuse_a, reuse_b)
  return __builtin_amdgcn_wmma_f32_16x16x32_bf16(false, a, false, b, (short)0, c,
                                                 false, false);
}

DEV float sigf(float x) { return 1.f / (1.f + __expf(-x)); }

// A-fragment (16x32, M x K), row-major source with leading dim `ld` (elems).
// lane<16: row=lane, k = kbase+0..15 contiguous; lane>=16: row=lane-16, k = kbase+16..31.
DEV v16bf load_a_frag(const __bf16* base, int ld, int row0, int kbase, int lane) {
  int m  = row0 + (lane & 15);
  int k0 = kbase + ((lane >> 4) << 4);
  const __bf16* p = base + (size_t)m * ld + k0;
  v8bf lo = *reinterpret_cast<const v8bf*>(p);
  v8bf hi = *reinterpret_cast<const v8bf*>(p + 8);
  v16bf f;
#pragma unroll
  for (int i = 0; i < 8; ++i) { f[i] = lo[i]; f[i + 8] = hi[i]; }
  return f;
}

// B-fragment (32x16, K x N) sourced from row-major W[N,K] (computing A*W^T).
// lane<16: n=n0+lane, k = kbase+0..15 contiguous; lane>=16: k = kbase+16..31.
DEV v16bf load_b_frag(const __bf16* base, int ld, int n0, int kbase, int lane) {
  int n  = n0 + (lane & 15);
  int k0 = kbase + ((lane >> 4) << 4);
  return *reinterpret_cast<const v16bf*>(base + (size_t)n * ld + k0);
}

// ---------------------------------------------------------------------------
// Conversion / gather kernels
// ---------------------------------------------------------------------------

__global__ __launch_bounds__(256)
void convert_pad_kernel(const float* __restrict__ in, __bf16* __restrict__ out,
                        int N, int Kin, int Kout) {
  int i = blockIdx.x * 256 + threadIdx.x;
  if (i >= N * Kout) return;
  int n = i / Kout, k = i - n * Kout;
  out[i] = (__bf16)((k < Kin) ? in[(size_t)n * Kin + k] : 0.f);
}

__global__ __launch_bounds__(256)
void convert_kernel(const float* __restrict__ in, __bf16* __restrict__ out, int n) {
  int i = blockIdx.x * 256 + threadIdx.x;
  if (i < n) out[i] = (__bf16)in[i];
}

// Embedding gather, time-major rows (row = t*16 + b), K padded 300 -> 320.
__global__ __launch_bounds__(320)
void embed_kernel(const int* __restrict__ tok, const float* __restrict__ emb,
                  __bf16* __restrict__ out, int T) {
  int row = blockIdx.x;      // t*16 + b
  int t = row >> 4;
  int b = row & 15;
  int col = threadIdx.x;     // 0..319
  int idx = tok[b * T + t];
  float v = (col < 300) ? emb[(size_t)idx * 300 + col] : 0.f;
  out[(size_t)row * 320 + col] = (__bf16)v;
}

// Vt[b][c][k] = Vv[b*64+k][c]  (per-batch transpose to bf16, ld=64)
__global__ __launch_bounds__(256)
void vtrans_kernel(const float* __restrict__ Vv, __bf16* __restrict__ Vt) {
  int i = blockIdx.x * 256 + threadIdx.x;          // b*65536 + c*64 + k
  int b = i >> 16, rem = i & 65535, c = rem >> 6, k = rem & 63;
  Vt[i] = (__bf16)Vv[(((size_t)b * 64 + k) << 10) + c];
}

// cat[row][1024 + c] = Q[row][c]
__global__ __launch_bounds__(256)
void qcat_kernel(const float* __restrict__ Q, __bf16* __restrict__ cat) {
  int i = blockIdx.x * 256 + threadIdx.x;          // row*1024 + c
  int row = i >> 10, c = i & 1023;
  cat[(size_t)row * 2048 + 1024 + c] = (__bf16)Q[i];
}

// ---------------------------------------------------------------------------
// Generic tiled WMMA GEMM: out[M,N] = A[M,K] * W[N,K]^T + bias[N]  (opt tanh)
// block = 128 (4 waves, one 16x16 n-tile each), grid = (ceil(N/64), M/16)
// ---------------------------------------------------------------------------
__global__ __launch_bounds__(128)
void gemm_bias_kernel(const __bf16* __restrict__ A, int lda,
                      const __bf16* __restrict__ W, int ldw,
                      const float* __restrict__ bias,
                      float* __restrict__ out, int ldo,
                      int K, int ntiles, int act) {
  int lane = threadIdx.x & 31;
  int nt = blockIdx.x * 4 + (threadIdx.x >> 5);
  int mt = blockIdx.y;
  if (nt >= ntiles) return;
  v8f acc = {};
  for (int k0 = 0; k0 < K; k0 += 32) {
    v16bf a = load_a_frag(A, lda, mt * 16, k0, lane);
    v16bf b = load_b_frag(W, ldw, nt * 16, k0, lane);
    acc = wmma_bf16(a, b, acc);
  }
  int colb = lane & 15;
  int n = nt * 16 + colb;
  float bs = bias[n];
#pragma unroll
  for (int r = 0; r < 8; ++r) {
    int m = mt * 16 + r + ((lane >> 4) << 3);
    float v = acc[r] + bs;
    if (act) v = tanhf(v);
    out[(size_t)m * ldo + n] = v;
  }
}

// ---------------------------------------------------------------------------
// Persistent bi-encoder LSTM. grid.x = 2 (fwd/bwd), 1024 threads = 32 waves.
// Wave w owns hidden columns [16w,16w+16): its own i/f/g/o gate tiles and a
// register-resident cell-state fragment. h kept in LDS as bf16 (A operand).
// ---------------------------------------------------------------------------
__global__ __launch_bounds__(1024)
void lstm_enc_kernel(const float* __restrict__ xg_f, const float* __restrict__ xg_b,
                     const __bf16* __restrict__ whh_f, const __bf16* __restrict__ whh_b,
                     float* __restrict__ enc_out, float* __restrict__ hT,
                     float* __restrict__ cT) {
  const int dir = blockIdx.x;
  const float*  xg  = dir ? xg_b : xg_f;        // [64*16, 2048] time-major
  const __bf16* whh = dir ? whh_b : whh_f;      // [2048, 512] bf16
  __shared__ __bf16 hbuf[16 * 512];
  const int tid = threadIdx.x;
  const int wave = tid >> 5, lane = tid & 31;
  const int cb = wave;                  // hidden column block
  const int colb = lane & 15;
  const int rbase = (lane >> 4) << 3;

  for (int i = tid; i < 16 * 512; i += 1024) hbuf[i] = (__bf16)0.f;
  __syncthreads();

  v8f c_frag = {};
  for (int t = 0; t < 64; ++t) {
    const int t_x = dir ? (63 - t) : t;
    v8f ai, af, ag, ao;
#pragma unroll
    for (int r = 0; r < 8; ++r) {
      const float* g = xg + (size_t)(t_x * 16 + r + rbase) * 2048 + cb * 16 + colb;
      ai[r] = g[0]; af[r] = g[512]; ag[r] = g[1024]; ao[r] = g[1536];
    }
    for (int k0 = 0; k0 < 512; k0 += 32) {
      v16bf a = load_a_frag(hbuf, 512, 0, k0, lane);
      ai = wmma_bf16(a, load_b_frag(whh, 512, 0 * 512 + cb * 16, k0, lane), ai);
      af = wmma_bf16(a, load_b_frag(whh, 512, 1 * 512 + cb * 16, k0, lane), af);
      ag = wmma_bf16(a, load_b_frag(whh, 512, 2 * 512 + cb * 16, k0, lane), ag);
      ao = wmma_bf16(a, load_b_frag(whh, 512, 3 * 512 + cb * 16, k0, lane), ao);
    }
    v8f hv;
#pragma unroll
    for (int r = 0; r < 8; ++r) {
      float iv = sigf(ai[r]), fv = sigf(af[r]);
      float gv = tanhf(ag[r]), ov = sigf(ao[r]);
      float c = fv * c_frag[r] + iv * gv;
      c_frag[r] = c;
      hv[r] = ov * tanhf(c);
    }
    __syncthreads();   // everyone done reading hbuf for this step
#pragma unroll
    for (int r = 0; r < 8; ++r) {
      int m = r + rbase;
      int col = cb * 16 + colb;
      hbuf[m * 512 + col] = (__bf16)hv[r];
      // enc_out: [b, t, 1024], bwd output re-reversed via t_x
      enc_out[((size_t)m * 64 + t_x) * 1024 + dir * 512 + col] = hv[r];
    }
    __syncthreads();
  }
#pragma unroll
  for (int r = 0; r < 8; ++r) {
    int m = r + rbase;
    int col = cb * 16 + colb;
    hT[((size_t)dir * 16 + m) * 512 + col] = (float)hbuf[m * 512 + col];
    cT[((size_t)dir * 16 + m) * 512 + col] = c_frag[r];
  }
}

// ---------------------------------------------------------------------------
// Persistent decoder LSTM, hidden 1024 (gates 4096). 1 block, 32 waves.
// Wave w owns column blocks {2w, 2w+1} -> 8 WMMA accumulators per step.
// h0/c0 interleave fwd/bwd encoder finals: ch even -> fwd[ch/2], odd -> bwd.
// ---------------------------------------------------------------------------
__global__ __launch_bounds__(1024)
void lstm_dec_kernel(const float* __restrict__ xg, const __bf16* __restrict__ whh,
                     const float* __restrict__ hT, const float* __restrict__ cT,
                     float* __restrict__ Q) {
  __shared__ __bf16 hbuf[16 * 1024];
  const int tid = threadIdx.x;
  const int wave = tid >> 5, lane = tid & 31;
  const int colb = lane & 15;
  const int rbase = (lane >> 4) << 3;

  for (int i = tid; i < 16 * 1024; i += 1024) {
    int b = i >> 10, ch = i & 1023;
    hbuf[i] = (__bf16)hT[(((size_t)(ch & 1) * 16 + b) << 9) + (ch >> 1)];
  }
  __syncthreads();

  v8f c_frag[2];
#pragma unroll
  for (int cbi = 0; cbi < 2; ++cbi) {
    int cb = wave * 2 + cbi;
#pragma unroll
    for (int r = 0; r < 8; ++r) {
      int m = r + rbase;
      int ch = cb * 16 + colb;
      c_frag[cbi][r] = cT[(((size_t)(ch & 1) * 16 + m) << 9) + (ch >> 1)];
    }
  }

  for (int t = 0; t < 64; ++t) {
    v8f acc[2][4];
#pragma unroll
    for (int cbi = 0; cbi < 2; ++cbi) {
      int cb = wave * 2 + cbi;
#pragma unroll
      for (int gate = 0; gate < 4; ++gate)
#pragma unroll
        for (int r = 0; r < 8; ++r)
          acc[cbi][gate][r] =
              xg[(size_t)(t * 16 + r + rbase) * 4096 + gate * 1024 + cb * 16 + colb];
    }
    for (int k0 = 0; k0 < 1024; k0 += 32) {
      v16bf a = load_a_frag(hbuf, 1024, 0, k0, lane);
#pragma unroll
      for (int cbi = 0; cbi < 2; ++cbi) {
        int cb = wave * 2 + cbi;
#pragma unroll
        for (int gate = 0; gate < 4; ++gate)
          acc[cbi][gate] = wmma_bf16(
              a, load_b_frag(whh, 1024, gate * 1024 + cb * 16, k0, lane),
              acc[cbi][gate]);
      }
    }
    __syncthreads();
#pragma unroll
    for (int cbi = 0; cbi < 2; ++cbi) {
      int cb = wave * 2 + cbi;
#pragma unroll
      for (int r = 0; r < 8; ++r) {
        float iv = sigf(acc[cbi][0][r]);
        float fv = sigf(acc[cbi][1][r]);
        float gv = tanhf(acc[cbi][2][r]);
        float ov = sigf(acc[cbi][3][r]);
        float c = fv * c_frag[cbi][r] + iv * gv;
        c_frag[cbi][r] = c;
        float h = ov * tanhf(c);
        int m = r + rbase;
        int col = cb * 16 + colb;
        hbuf[m * 1024 + col] = (__bf16)h;
        Q[((size_t)m * 64 + t) * 1024 + col] = h;   // Q: [b, t, 1024]
      }
    }
    __syncthreads();
  }
}

// ---------------------------------------------------------------------------
// Attention
// ---------------------------------------------------------------------------

// scores[b,q,k] = Q[b,q,:].K[b,k,:]  (+ PAD mask). grid(4,4,16), 1 wave.
__global__ __launch_bounds__(32)
void attn_scores_kernel(const __bf16* __restrict__ Qb, const __bf16* __restrict__ Kb,
                        const int* __restrict__ src, float* __restrict__ S) {
  const int qt = blockIdx.x, kt = blockIdx.y, b = blockIdx.z;
  const int lane = threadIdx.x;
  const __bf16* Abase = Qb + ((size_t)b << 16);  // 64*1024
  const __bf16* Bbase = Kb + ((size_t)b << 16);
  v8f acc = {};
  for (int k0 = 0; k0 < 1024; k0 += 32) {
    v16bf a  = load_a_frag(Abase, 1024, qt * 16, k0, lane);
    v16bf bb = load_b_frag(Bbase, 1024, kt * 16, k0, lane);
    acc = wmma_bf16(a, bb, acc);
  }
  const int colb = lane & 15;
  const int kcol = kt * 16 + colb;
  const bool masked = (src[b * 64 + kcol] == 0);  // PAD == 0
#pragma unroll
  for (int r = 0; r < 8; ++r) {
    int q = qt * 16 + r + ((lane >> 4) << 3);
    S[((size_t)b * 64 + q) * 64 + kcol] = masked ? -1e30f : acc[r];
  }
}

// row softmax over 64, one wave per (b,q) row; writes bf16 probs.
__global__ __launch_bounds__(32)
void softmax_kernel(const float* __restrict__ S, __bf16* __restrict__ P) {
  const int row = blockIdx.x;
  const int lane = threadIdx.x;
  float v0 = S[row * 64 + lane];
  float v1 = S[row * 64 + 32 + lane];
  float m = fmaxf(v0, v1);
#pragma unroll
  for (int o = 16; o > 0; o >>= 1) m = fmaxf(m, __shfl_xor(m, o, 32));
  float e0 = __expf(v0 - m), e1 = __expf(v1 - m);
  float s = e0 + e1;
#pragma unroll
  for (int o = 16; o > 0; o >>= 1) s += __shfl_xor(s, o, 32);
  float inv = 1.f / s;
  P[row * 64 + lane]      = (__bf16)(e0 * inv);
  P[row * 64 + 32 + lane] = (__bf16)(e1 * inv);
}

// ctx = P @ V ; writes bf16 straight into left half of cat. grid(64,4,16).
__global__ __launch_bounds__(32)
void attn_ctx_kernel(const __bf16* __restrict__ P, const __bf16* __restrict__ Vt,
                     __bf16* __restrict__ cat) {
  const int ct = blockIdx.x, qt = blockIdx.y, b = blockIdx.z;
  const int lane = threadIdx.x;
  const __bf16* Abase = P + ((size_t)b << 12);   // 64*64, ld 64
  const __bf16* Bbase = Vt + ((size_t)b << 16);  // 1024*64, ld 64
  v8f acc = {};
  for (int k0 = 0; k0 < 64; k0 += 32) {
    v16bf a  = load_a_frag(Abase, 64, qt * 16, k0, lane);
    v16bf bb = load_b_frag(Bbase, 64, ct * 16, k0, lane);
    acc = wmma_bf16(a, bb, acc);
  }
  const int colb = lane & 15;
  const int c = ct * 16 + colb;
#pragma unroll
  for (int r = 0; r < 8; ++r) {
    int q = qt * 16 + r + ((lane >> 4) << 3);
    cat[((size_t)b * 64 + q) * 2048 + c] = (__bf16)acc[r];
  }
}

// ---------------------------------------------------------------------------
// Host driver
// ---------------------------------------------------------------------------
extern "C" void kernel_launch(void* const* d_in, const int* in_sizes, int n_in,
                              void* d_out, int out_size, void* d_ws, size_t ws_size,
                              hipStream_t stream) {
  (void)in_sizes; (void)n_in; (void)out_size; (void)ws_size;
  const int*   source = (const int*)d_in[0];
  const int*   prev   = (const int*)d_in[1];
  const float* emb    = (const float*)d_in[2];
  const float* efWih  = (const float*)d_in[3];
  const float* efWhh  = (const float*)d_in[4];
  const float* efB    = (const float*)d_in[5];
  const float* ebWih  = (const float*)d_in[6];
  const float* ebWhh  = (const float*)d_in[7];
  const float* ebB    = (const float*)d_in[8];
  const float* dWih   = (const float*)d_in[9];
  const float* dWhh   = (const float*)d_in[10];
  const float* dB     = (const float*)d_in[11];
  const float* kW     = (const float*)d_in[12];
  const float* kB     = (const float*)d_in[13];
  const float* vW     = (const float*)d_in[14];
  const float* vB     = (const float*)d_in[15];
  const float* fc1W   = (const float*)d_in[16];
  const float* fc1B   = (const float*)d_in[17];
  const float* fc2W   = (const float*)d_in[18];
  const float* fc2B   = (const float*)d_in[19];
  float* out = (float*)d_out;

  char* p = (char*)d_ws;
  auto alloc = [&](size_t bytes) -> void* {
    void* r = (void*)p;
    p += (bytes + 255) & ~(size_t)255;
    return r;
  };

  __bf16* xs_src  = (__bf16*)alloc((size_t)1024 * 320 * 2);
  __bf16* xs_prev = (__bf16*)alloc((size_t)1024 * 320 * 2);
  __bf16* wihF    = (__bf16*)alloc((size_t)2048 * 320 * 2);
  __bf16* wihB    = (__bf16*)alloc((size_t)2048 * 320 * 2);
  __bf16* wihD    = (__bf16*)alloc((size_t)4096 * 320 * 2);
  __bf16* whhF    = (__bf16*)alloc((size_t)2048 * 512 * 2);
  __bf16* whhB    = (__bf16*)alloc((size_t)2048 * 512 * 2);
  __bf16* whhD    = (__bf16*)alloc((size_t)4096 * 1024 * 2);
  __bf16* kWb     = (__bf16*)alloc((size_t)1024 * 1024 * 2);
  __bf16* vWb     = (__bf16*)alloc((size_t)1024 * 1024 * 2);
  __bf16* fc1Wb   = (__bf16*)alloc((size_t)1024 * 2048 * 2);
  __bf16* fc2Wb   = (__bf16*)alloc((size_t)10000 * 1024 * 2);
  float*  xgF     = (float*)alloc((size_t)1024 * 2048 * 4);
  float*  xgB     = (float*)alloc((size_t)1024 * 2048 * 4);
  float*  xgD     = (float*)alloc((size_t)1024 * 4096 * 4);
  float*  encOut  = (float*)alloc((size_t)1024 * 1024 * 4);
  __bf16* encOutB = (__bf16*)alloc((size_t)1024 * 1024 * 2);
  float*  hT      = (float*)alloc((size_t)2 * 16 * 512 * 4);
  float*  cT      = (float*)alloc((size_t)2 * 16 * 512 * 4);
  float*  Qf      = (float*)alloc((size_t)1024 * 1024 * 4);
  __bf16* Qb      = (__bf16*)alloc((size_t)1024 * 1024 * 2);
  float*  Kf      = (float*)alloc((size_t)1024 * 1024 * 4);
  __bf16* Kb      = (__bf16*)alloc((size_t)1024 * 1024 * 2);
  float*  Vf      = (float*)alloc((size_t)1024 * 1024 * 4);
  __bf16* Vt      = (__bf16*)alloc((size_t)1024 * 1024 * 2);
  float*  S       = (float*)alloc((size_t)16 * 64 * 64 * 4);
  __bf16* Pm      = (__bf16*)alloc((size_t)16 * 64 * 64 * 2);
  __bf16* catB    = (__bf16*)alloc((size_t)1024 * 2048 * 2);
  float*  fc1f    = (float*)alloc((size_t)1024 * 1024 * 4);
  __bf16* fc1b16  = (__bf16*)alloc((size_t)1024 * 1024 * 2);

  auto nb = [](int n) { return (n + 255) / 256; };

  // Weight conversions (fp32 -> bf16, Wih padded K 300->320)
  convert_pad_kernel<<<nb(2048 * 320), 256, 0, stream>>>(efWih, wihF, 2048, 300, 320);
  convert_pad_kernel<<<nb(2048 * 320), 256, 0, stream>>>(ebWih, wihB, 2048, 300, 320);
  convert_pad_kernel<<<nb(4096 * 320), 256, 0, stream>>>(dWih,  wihD, 4096, 300, 320);
  convert_kernel<<<nb(2048 * 512), 256, 0, stream>>>(efWhh, whhF, 2048 * 512);
  convert_kernel<<<nb(2048 * 512), 256, 0, stream>>>(ebWhh, whhB, 2048 * 512);
  convert_kernel<<<nb(4096 * 1024), 256, 0, stream>>>(dWhh, whhD, 4096 * 1024);
  convert_kernel<<<nb(1024 * 1024), 256, 0, stream>>>(kW, kWb, 1024 * 1024);
  convert_kernel<<<nb(1024 * 1024), 256, 0, stream>>>(vW, vWb, 1024 * 1024);
  convert_kernel<<<nb(1024 * 2048), 256, 0, stream>>>(fc1W, fc1Wb, 1024 * 2048);
  convert_kernel<<<nb(10000 * 1024), 256, 0, stream>>>(fc2W, fc2Wb, 10000 * 1024);

  // Embedding gathers (time-major rows)
  embed_kernel<<<1024, 320, 0, stream>>>(source, emb, xs_src, 64);
  embed_kernel<<<1024, 320, 0, stream>>>(prev,   emb, xs_prev, 64);

  // Input-gate pre-projections: xg = xs @ Wih^T + b
  gemm_bias_kernel<<<dim3(32, 64), 128, 0, stream>>>(xs_src, 320, wihF, 320, efB,
                                                     xgF, 2048, 320, 128, 0);
  gemm_bias_kernel<<<dim3(32, 64), 128, 0, stream>>>(xs_src, 320, wihB, 320, ebB,
                                                     xgB, 2048, 320, 128, 0);
  gemm_bias_kernel<<<dim3(64, 64), 128, 0, stream>>>(xs_prev, 320, wihD, 320, dB,
                                                     xgD, 4096, 320, 256, 0);

  // Bidirectional encoder scan (persistent, register-resident cell state)
  lstm_enc_kernel<<<2, 1024, 0, stream>>>(xgF, xgB, whhF, whhB, encOut, hT, cT);

  // K/V projections of enc_out
  convert_kernel<<<nb(1024 * 1024), 256, 0, stream>>>(encOut, encOutB, 1024 * 1024);
  gemm_bias_kernel<<<dim3(16, 64), 128, 0, stream>>>(encOutB, 1024, kWb, 1024, kB,
                                                     Kf, 1024, 1024, 64, 0);
  gemm_bias_kernel<<<dim3(16, 64), 128, 0, stream>>>(encOutB, 1024, vWb, 1024, vB,
                                                     Vf, 1024, 1024, 64, 0);
  convert_kernel<<<nb(1024 * 1024), 256, 0, stream>>>(Kf, Kb, 1024 * 1024);
  vtrans_kernel<<<4096, 256, 0, stream>>>(Vf, Vt);

  // Decoder scan (hidden 1024, init from interleaved encoder finals)
  lstm_dec_kernel<<<1, 1024, 0, stream>>>(xgD, whhD, hT, cT, Qf);
  convert_kernel<<<nb(1024 * 1024), 256, 0, stream>>>(Qf, Qb, 1024 * 1024);

  // Attention
  attn_scores_kernel<<<dim3(4, 4, 16), 32, 0, stream>>>(Qb, Kb, source, S);
  softmax_kernel<<<1024, 32, 0, stream>>>(S, Pm);
  attn_ctx_kernel<<<dim3(64, 4, 16), 32, 0, stream>>>(Pm, Vt, catB);
  qcat_kernel<<<4096, 256, 0, stream>>>(Qf, catB);

  // fc1 (tanh) then fc2 -> d_out
  gemm_bias_kernel<<<dim3(16, 64), 128, 0, stream>>>(catB, 2048, fc1Wb, 2048, fc1B,
                                                     fc1f, 1024, 2048, 64, 1);
  convert_kernel<<<nb(1024 * 1024), 256, 0, stream>>>(fc1f, fc1b16, 1024 * 1024);
  gemm_bias_kernel<<<dim3(157, 64), 128, 0, stream>>>(fc1b16, 1024, fc2Wb, 1024, fc2B,
                                                      out, 10000, 1024, 625, 0);
}
